// MixtralForCausalLM_2087354105881
// MI455X (gfx1250) — compile-verified, matrix operand order checked by reading, not run on hardware
//
#include <hip/hip_runtime.h>
#include <hip/hip_bf16.h>
#include <math.h>

#define T_TOK 8192
#define HDIM  2048
#define IDIM  7168
#define NEXP  8
#define TOPK  2

#define BM 64
#define BN 128
#define BK 32

typedef __bf16 bf16_t;
typedef __attribute__((ext_vector_type(16))) __bf16 v16bf;
typedef __attribute__((ext_vector_type(8)))  __bf16 v8bf;
typedef __attribute__((ext_vector_type(8)))  float  v8f;
typedef __attribute__((ext_vector_type(4)))  float  v4f;
typedef __attribute__((ext_vector_type(4)))  int    v4i;

union FragU { v16bf v; v8bf h[2]; };
union PackU { bf16_t h[2]; unsigned u; };

#if __has_builtin(__builtin_amdgcn_global_load_async_to_lds_b128)
#define HAS_ASYNC_LDS 1
#else
#define HAS_ASYNC_LDS 0
#endif

__device__ __forceinline__ void async_wait0() {
#if __has_builtin(__builtin_amdgcn_s_wait_asynccnt)
    __builtin_amdgcn_s_wait_asynccnt(0);
#else
    asm volatile("s_wait_asynccnt 0" ::: "memory");
#endif
}

// async copy of 16 bytes (one v8bf chunk) global -> LDS, ASYNCcnt-tracked
__device__ __forceinline__ void async_copy16(const bf16_t* g, bf16_t* l) {
#if HAS_ASYNC_LDS
    __builtin_amdgcn_global_load_async_to_lds_b128(
        (v4i*)(void*)g, (v4i*)(void*)l, 0, 0);
#endif
}

// ---------------- small helper kernels ----------------

__global__ void zero_counts_kernel(int* counts) {
    if (threadIdx.x < NEXP) counts[threadIdx.x] = 0;
}

__global__ void zero_out_kernel(float* __restrict__ out) {
    size_t i = (size_t)blockIdx.x * blockDim.x + threadIdx.x;
    v4f z = {0.f, 0.f, 0.f, 0.f};
    ((v4f*)out)[i] = z;
}

__global__ void cvt_x_kernel(const float* __restrict__ x, bf16_t* __restrict__ xb) {
    size_t i = ((size_t)blockIdx.x * blockDim.x + threadIdx.x) * 8;
    v4f a = *(const v4f*)(x + i);
    v4f b = *(const v4f*)(x + i + 4);
    v8bf o;
    o[0] = (bf16_t)a[0]; o[1] = (bf16_t)a[1]; o[2] = (bf16_t)a[2]; o[3] = (bf16_t)a[3];
    o[4] = (bf16_t)b[0]; o[5] = (bf16_t)b[1]; o[6] = (bf16_t)b[2]; o[7] = (bf16_t)b[3];
    *(v8bf*)(xb + i) = o;
}

// ---------------- gating / routing ----------------

__global__ __launch_bounds__(256)
void gate_kernel(const float* __restrict__ x, const float* __restrict__ wg,
                 int* __restrict__ counts, int* __restrict__ tids,
                 float* __restrict__ sw) {
    const int t    = blockIdx.x;
    const int lane = threadIdx.x & 31;
    const int w    = threadIdx.x >> 5;      // expert handled by this wave
    const float* xr = x + (size_t)t * HDIM;

    float acc = 0.f;
    for (int h = lane; h < HDIM; h += 32)
        acc += xr[h] * wg[h * NEXP + w];
    for (int off = 16; off; off >>= 1)
        acc += __shfl_xor(acc, off, 32);

    __shared__ float slog[NEXP];
    if (lane == 0) slog[w] = acc;
    __syncthreads();

    if (threadIdx.x == 0) {
        float m = slog[0];
        #pragma unroll
        for (int i = 1; i < NEXP; ++i) m = fmaxf(m, slog[i]);
        float pe[NEXP];
        #pragma unroll
        for (int i = 0; i < NEXP; ++i) pe[i] = __expf(slog[i] - m);
        // top-2, first-occurrence tie-break (matches jax.lax.top_k)
        int i0 = 0;
        #pragma unroll
        for (int i = 1; i < NEXP; ++i) if (pe[i] > pe[i0]) i0 = i;
        int i1 = -1;
        #pragma unroll
        for (int i = 0; i < NEXP; ++i) {
            if (i == i0) continue;
            if (i1 < 0 || pe[i] > pe[i1]) i1 = i;
        }
        float s  = pe[i0] + pe[i1];
        float w0 = pe[i0] / s;
        float w1 = pe[i1] / s;
        int s0 = atomicAdd(&counts[i0], 1);
        tids[i0 * T_TOK + s0] = t;
        sw  [i0 * T_TOK + s0] = w0;
        int s1 = atomicAdd(&counts[i1], 1);
        tids[i1 * T_TOK + s1] = t;
        sw  [i1 * T_TOK + s1] = w1;
    }
}

__global__ void scan_kernel(const int* __restrict__ counts, int* __restrict__ base) {
    if (threadIdx.x == 0) {
        int b = 0;
        #pragma unroll
        for (int e = 0; e < NEXP; ++e) { base[e] = b; b += counts[e]; }
    }
}

// pack a (k, k+1) bf16 pair and store as one dword into LDS laid out [n][k]
__device__ __forceinline__ void stage_pair(bf16_t* __restrict__ dst, int nq, int k0,
                                           v4f r0, v4f r1) {
    #pragma unroll
    for (int j = 0; j < 4; ++j) {
        PackU p;
        p.h[0] = (bf16_t)r0[j];
        p.h[1] = (bf16_t)r1[j];
        *(unsigned*)&dst[(nq * 4 + j) * BK + k0] = p.u;
    }
}

__device__ __forceinline__ v16bf frag_ld(const bf16_t* __restrict__ s, int r, int lhi) {
    FragU u;
    u.h[0] = *(const v8bf*)(&s[r * BK + lhi * 8]);
    u.h[1] = *(const v8bf*)(&s[r * BK + 16 + lhi * 8]);
    return u.v;
}

// ---------------- GEMM1: act = silu(x@w1) * (x@w3), bf16 out ----------------

__global__ __launch_bounds__(256)
void gemm1_kernel(const bf16_t* __restrict__ xb,
                  const float* __restrict__ w1,
                  const float* __restrict__ w3,
                  const int* __restrict__ counts, const int* __restrict__ basep,
                  const int* __restrict__ tids,
                  bf16_t* __restrict__ act) {
    const int e     = blockIdx.z;
    const int count = counts[e];
    const int m0    = blockIdx.y * BM;
    if (m0 >= count) return;
    const int n0    = blockIdx.x * BN;
    const int base  = basep[e];
    const int tid   = threadIdx.x;

    __shared__ int s_tid[BM];
    __shared__ __attribute__((aligned(16))) bf16_t sA [2][BM * BK];
    __shared__ __attribute__((aligned(16))) bf16_t sB1[2][BN * BK];
    __shared__ __attribute__((aligned(16))) bf16_t sB3[2][BN * BK];

    if (tid < BM) {
        int slot = m0 + tid;
        s_tid[tid] = (slot < count) ? tids[e * T_TOK + slot] : 0;
    }
    __syncthreads();

    const int wv   = tid >> 5;
    const int lane = tid & 31;
    const int wm   = wv & 1;       // 2 wave rows
    const int wn   = wv >> 1;      // 4 wave cols
    const int lrow = lane & 15;
    const int lhi  = lane >> 4;

    v8f zf = {0.f,0.f,0.f,0.f,0.f,0.f,0.f,0.f};
    v8f accG[2][2], accU[2][2];
    #pragma unroll
    for (int i = 0; i < 2; ++i)
        #pragma unroll
        for (int j = 0; j < 2; ++j) { accG[i][j] = zf; accU[i][j] = zf; }

    // A staging: thread -> (row, 8-wide column chunk); token hoisted out of k-loop
    const int arow = tid >> 2;
    const int acol = (tid & 3) * 8;
    const int tok  = s_tid[arow];
    const bf16_t* aptr = xb + (size_t)tok * HDIM + acol;
    bf16_t* ldsA[2] = { &sA[0][arow * BK + acol], &sA[1][arow * BK + acol] };

    // B staging: 2 groups of (k-pair, n-quad)
    int kp0[2], nqv[2];
    #pragma unroll
    for (int g = 0; g < 2; ++g) {
        int lin = tid + g * 256;       // [0, 512)
        kp0[g] = (lin >> 5) * 2;       // k0 = 0,2,...,30
        nqv[g] = lin & 31;             // n quad
    }
    const float* w1b = w1 + (size_t)e * HDIM * IDIM + n0;
    const float* w3b = w3 + (size_t)e * HDIM * IDIM + n0;

    // ---- prologue: load k-step 0 and stage into buffer 0 ----
#if HAS_ASYNC_LDS
    async_copy16(aptr, ldsA[0]);
#else
    v8bf aReg = *(const v8bf*)(aptr);
#endif
    v4f b1r[2][2], b3r[2][2];
    #pragma unroll
    for (int g = 0; g < 2; ++g) {
        size_t o = (size_t)kp0[g] * IDIM + nqv[g] * 4;
        b1r[g][0] = *(const v4f*)(w1b + o);
        b1r[g][1] = *(const v4f*)(w1b + o + IDIM);
        b3r[g][0] = *(const v4f*)(w3b + o);
        b3r[g][1] = *(const v4f*)(w3b + o + IDIM);
    }
#if !HAS_ASYNC_LDS
    *(v8bf*)ldsA[0] = aReg;
#endif
    #pragma unroll
    for (int g = 0; g < 2; ++g) {
        stage_pair(sB1[0], nqv[g], kp0[g], b1r[g][0], b1r[g][1]);
        stage_pair(sB3[0], nqv[g], kp0[g], b3r[g][0], b3r[g][1]);
    }
#if HAS_ASYNC_LDS
    async_wait0();
#endif
    __syncthreads();

    int s = 0;
    for (int kk = 0; kk < HDIM; kk += BK) {
        const bool hasNext = (kk + BK) < HDIM;
        // prefetch next k-step while this stage computes
        if (hasNext) {
#if HAS_ASYNC_LDS
            async_copy16(aptr + kk + BK, ldsA[s ^ 1]);
#else
            aReg = *(const v8bf*)(aptr + kk + BK);
#endif
            #pragma unroll
            for (int g = 0; g < 2; ++g) {
                size_t o = (size_t)(kk + BK + kp0[g]) * IDIM + nqv[g] * 4;
                b1r[g][0] = *(const v4f*)(w1b + o);
                b1r[g][1] = *(const v4f*)(w1b + o + IDIM);
                b3r[g][0] = *(const v4f*)(w3b + o);
                b3r[g][1] = *(const v4f*)(w3b + o + IDIM);
            }
        }

        v16bf a[2], bf1[2], bf3[2];
        #pragma unroll
        for (int mi = 0; mi < 2; ++mi)
            a[mi] = frag_ld(sA[s], wm * 32 + mi * 16 + lrow, lhi);
        #pragma unroll
        for (int ni = 0; ni < 2; ++ni) {
            int r = wn * 32 + ni * 16 + lrow;
            bf1[ni] = frag_ld(sB1[s], r, lhi);
            bf3[ni] = frag_ld(sB3[s], r, lhi);
        }
        #pragma unroll
        for (int mi = 0; mi < 2; ++mi)
            #pragma unroll
            for (int ni = 0; ni < 2; ++ni) {
                accG[mi][ni] = __builtin_amdgcn_wmma_f32_16x16x32_bf16(
                    false, a[mi], false, bf1[ni], (short)0, accG[mi][ni], false, false);
                accU[mi][ni] = __builtin_amdgcn_wmma_f32_16x16x32_bf16(
                    false, a[mi], false, bf3[ni], (short)0, accU[mi][ni], false, false);
            }

        if (hasNext) {
            int nx = s ^ 1;
#if !HAS_ASYNC_LDS
            *(v8bf*)ldsA[nx] = aReg;
#endif
            #pragma unroll
            for (int g = 0; g < 2; ++g) {
                stage_pair(sB1[nx], nqv[g], kp0[g], b1r[g][0], b1r[g][1]);
                stage_pair(sB3[nx], nqv[g], kp0[g], b3r[g][0], b3r[g][1]);
            }
        }
#if HAS_ASYNC_LDS
        async_wait0();
#endif
        __syncthreads();
        s ^= 1;
    }

    // epilogue: silu(g)*u -> bf16 act; fast path for full tiles
    const bool full = (m0 + BM) <= count;
    #pragma unroll
    for (int mi = 0; mi < 2; ++mi) {
        #pragma unroll
        for (int r = 0; r < 8; ++r) {
            int mrow = wm * 32 + mi * 16 + r + lhi * 8;
            int slot = m0 + mrow;
            bf16_t* rowp = act + (size_t)(base + slot) * IDIM;
            if (full || slot < count) {
                #pragma unroll
                for (int ni = 0; ni < 2; ++ni) {
                    int ncol = n0 + wn * 32 + ni * 16 + lrow;
                    float g = accG[mi][ni][r];
                    float u = accU[mi][ni][r];
                    float sg = g * __builtin_amdgcn_rcpf(1.f + __expf(-g));
                    rowp[ncol] = (bf16_t)(sg * u);
                }
            }
        }
    }
}

// ---------------- GEMM2: out += w(slot) * (act @ w2) ----------------

__global__ __launch_bounds__(256)
void gemm2_kernel(const bf16_t* __restrict__ act,
                  const float* __restrict__ w2,
                  const int* __restrict__ counts, const int* __restrict__ basep,
                  const int* __restrict__ tids, const float* __restrict__ sw,
                  float* __restrict__ out) {
    const int e     = blockIdx.z;
    const int count = counts[e];
    const int m0    = blockIdx.y * BM;
    if (m0 >= count) return;
    const int n0    = blockIdx.x * BN;
    const int base  = basep[e];
    const int tid   = threadIdx.x;

    __shared__ int   s_tok[BM];
    __shared__ float s_w[BM];
    __shared__ __attribute__((aligned(16))) bf16_t sA[2][BM * BK];
    __shared__ __attribute__((aligned(16))) bf16_t sB[2][BN * BK];

    if (tid < BM) {
        int slot = m0 + tid;
        bool v = slot < count;
        s_tok[tid] = v ? tids[e * T_TOK + slot] : 0;
        s_w[tid]   = v ? sw[e * T_TOK + slot]   : 0.f;
    }
    __syncthreads();

    const int wv   = tid >> 5;
    const int lane = tid & 31;
    const int wm   = wv & 1;
    const int wn   = wv >> 1;
    const int lrow = lane & 15;
    const int lhi  = lane >> 4;

    v8f zf = {0.f,0.f,0.f,0.f,0.f,0.f,0.f,0.f};
    v8f acc[2][2];
    #pragma unroll
    for (int i = 0; i < 2; ++i)
        #pragma unroll
        for (int j = 0; j < 2; ++j) acc[i][j] = zf;

    const int arow  = tid >> 2;
    const int acol  = (tid & 3) * 8;
    const int aslot = m0 + arow;
    const int crow  = base + ((aslot < count) ? aslot : (count - 1));
    const bf16_t* aptr = act + (size_t)crow * IDIM + acol;
    bf16_t* ldsA[2] = { &sA[0][arow * BK + acol], &sA[1][arow * BK + acol] };

    int kp0[2], nqv[2];
    #pragma unroll
    for (int g = 0; g < 2; ++g) {
        int lin = tid + g * 256;
        kp0[g] = (lin >> 5) * 2;
        nqv[g] = lin & 31;
    }
    const float* w2b = w2 + (size_t)e * IDIM * HDIM + n0;

    // ---- prologue ----
#if HAS_ASYNC_LDS
    async_copy16(aptr, ldsA[0]);
#else
    v8bf aReg = *(const v8bf*)(aptr);
#endif
    v4f br[2][2];
    #pragma unroll
    for (int g = 0; g < 2; ++g) {
        size_t o = (size_t)kp0[g] * HDIM + nqv[g] * 4;
        br[g][0] = *(const v4f*)(w2b + o);
        br[g][1] = *(const v4f*)(w2b + o + HDIM);
    }
#if !HAS_ASYNC_LDS
    *(v8bf*)ldsA[0] = aReg;
#endif
    #pragma unroll
    for (int g = 0; g < 2; ++g)
        stage_pair(sB[0], nqv[g], kp0[g], br[g][0], br[g][1]);
#if HAS_ASYNC_LDS
    async_wait0();
#endif
    __syncthreads();

    int s = 0;
    for (int kk = 0; kk < IDIM; kk += BK) {
        const bool hasNext = (kk + BK) < IDIM;
        if (hasNext) {
#if HAS_ASYNC_LDS
            async_copy16(aptr + kk + BK, ldsA[s ^ 1]);
#else
            aReg = *(const v8bf*)(aptr + kk + BK);
#endif
            #pragma unroll
            for (int g = 0; g < 2; ++g) {
                size_t o = (size_t)(kk + BK + kp0[g]) * HDIM + nqv[g] * 4;
                br[g][0] = *(const v4f*)(w2b + o);
                br[g][1] = *(const v4f*)(w2b + o + HDIM);
            }
        }

        v16bf a[2], bfr[2];
        #pragma unroll
        for (int mi = 0; mi < 2; ++mi)
            a[mi] = frag_ld(sA[s], wm * 32 + mi * 16 + lrow, lhi);
        #pragma unroll
        for (int ni = 0; ni < 2; ++ni)
            bfr[ni] = frag_ld(sB[s], wn * 32 + ni * 16 + lrow, lhi);
        #pragma unroll
        for (int mi = 0; mi < 2; ++mi)
            #pragma unroll
            for (int ni = 0; ni < 2; ++ni)
                acc[mi][ni] = __builtin_amdgcn_wmma_f32_16x16x32_bf16(
                    false, a[mi], false, bfr[ni], (short)0, acc[mi][ni], false, false);

        if (hasNext) {
            int nx = s ^ 1;
#if !HAS_ASYNC_LDS
            *(v8bf*)ldsA[nx] = aReg;
#endif
            #pragma unroll
            for (int g = 0; g < 2; ++g)
                stage_pair(sB[nx], nqv[g], kp0[g], br[g][0], br[g][1]);
        }
#if HAS_ASYNC_LDS
        async_wait0();
#endif
        __syncthreads();
        s ^= 1;
    }

    // epilogue: weight * value scattered to out with fp32 atomics
    const bool full = (m0 + BM) <= count;
    #pragma unroll
    for (int mi = 0; mi < 2; ++mi) {
        #pragma unroll
        for (int r = 0; r < 8; ++r) {
            int mrow = wm * 32 + mi * 16 + r + lhi * 8;
            int slot = m0 + mrow;
            if (full || slot < count) {
                int   tokr = s_tok[mrow];
                float wgt  = s_w[mrow];
                float* rowp = out + (size_t)tokr * HDIM;
                #pragma unroll
                for (int ni = 0; ni < 2; ++ni) {
                    int ncol = n0 + wn * 32 + ni * 16 + lrow;
                    __hip_atomic_fetch_add(&rowp[ncol], acc[mi][ni][r] * wgt,
                                           __ATOMIC_RELAXED, __HIP_MEMORY_SCOPE_AGENT);
                }
            }
        }
    }
}

// ---------------- host launcher ----------------

extern "C" void kernel_launch(void* const* d_in, const int* in_sizes, int n_in,
                              void* d_out, int out_size, void* d_ws, size_t ws_size,
                              hipStream_t stream) {
    const float* x  = (const float*)d_in[0];
    const float* wg = (const float*)d_in[1];
    const float* w1 = (const float*)d_in[2];
    const float* w3 = (const float*)d_in[3];
    const float* w2 = (const float*)d_in[4];
    float* out = (float*)d_out;

    char* ws = (char*)d_ws;
    size_t off = 0;
    bf16_t* xb     = (bf16_t*)(ws + off); off += (size_t)T_TOK * HDIM * sizeof(bf16_t);
    int*    counts = (int*)   (ws + off); off += 256;
    int*    basep  = (int*)   (ws + off); off += 256;
    int*    tids   = (int*)   (ws + off); off += (size_t)NEXP * T_TOK * sizeof(int);
    float*  swt    = (float*) (ws + off); off += (size_t)NEXP * T_TOK * sizeof(float);
    bf16_t* act    = (bf16_t*)(ws + off); off += (size_t)TOPK * T_TOK * IDIM * sizeof(bf16_t);

    zero_counts_kernel<<<1, 32, 0, stream>>>(counts);
    zero_out_kernel<<<(T_TOK * HDIM / 4) / 256, 256, 0, stream>>>(out);
    cvt_x_kernel<<<(T_TOK * HDIM / 8) / 256, 256, 0, stream>>>(x, xb);
    gate_kernel<<<T_TOK, 256, 0, stream>>>(x, wg, counts, tids, swt);
    scan_kernel<<<1, 1, 0, stream>>>(counts, basep);

    dim3 g1(IDIM / BN, T_TOK / BM, NEXP);
    gemm1_kernel<<<g1, 256, 0, stream>>>(xb, w1, w3, counts, basep, tids, act);

    dim3 g2(HDIM / BN, T_TOK / BM, NEXP);
    gemm2_kernel<<<g2, 256, 0, stream>>>(act, w2, counts, basep, tids, swt, out);
}
